// vnet1D_inpaint_30829275250802
// MI455X (gfx1250) — matrix operand order repaired and checked by reading, not compiled, not run on hardware
//
#include <hip/hip_runtime.h>
#include <hip/hip_bf16.h>

// ---------------------------------------------------------------------------
// CDNA5 (gfx1250) implementation of the 1-D V-Net reference.
//  * convs = im2col GEMMs on v_wmma_f32_16x16x32_f16 (f16 in, f32 accumulate),
//    with fragment-friendly LDS layout (contiguous b128 ds loads, no per-elem
//    branching).
//  * instance-norm rows staged into LDS via the Tensor Data Mover
//    (tensor_load_to_lds + s_wait_tensorcnt), 6-arg builtin flavor.
//  * 134 MB pairwise-distance output is the HBM-bound tail (~6 us @ 23.3 TB/s).
// ---------------------------------------------------------------------------

typedef __attribute__((ext_vector_type(16))) _Float16 v16h;
typedef __attribute__((ext_vector_type(8)))  _Float16 v8h;
typedef __attribute__((ext_vector_type(8)))  float    v8f;
typedef unsigned int u32;
typedef __attribute__((ext_vector_type(4))) u32 v4u;
typedef __attribute__((ext_vector_type(8))) int v8i;
typedef __attribute__((ext_vector_type(4))) int v4i;

#define NB     8
#define NFULL  2048
#define PTILE  128
#define HSTEP  0.1f
#define DC_    3.79f
#define BIG_   10000000000.0f

static inline int ceil_div_i(int a, int b) { return (a + b - 1) / b; }

// ---------------------------------------------------------------------------
// TDM helpers: 1-D row copy (nelem f32, contiguous) global -> LDS.
// D# packing per CDNA5 ISA ch.8: group0 = {count|flags, lds_addr, gaddr_lo,
// gaddr_hi|type=2}; group1 carries data_size / dims / strides.
// ---------------------------------------------------------------------------
__device__ inline unsigned lds_off_u32(const void* p)
{
    // generic pointer to LDS: low 32 bits are the LDS byte offset
    return (unsigned)(unsigned long long)(uintptr_t)p;
}

__device__ inline void tdm_load_row_f32(unsigned lds_off, const float* gsrc, int nelem)
{
    unsigned long long ga = (unsigned long long)(uintptr_t)gsrc;
    v4u g0;
    g0[0] = 1u;                                            // count=1 (user desc)
    g0[1] = lds_off;                                       // lds_addr
    g0[2] = (u32)(ga & 0xffffffffu);                       // global_addr[31:0]
    g0[3] = (u32)((ga >> 32) & 0x1ffffffu) | (2u << 30);   // addr[56:32] | type=2
    u32 ne = (u32)nelem;
    v8i g1;
    g1[0] = (int)(2u << 16);                 // data_size=2 (4 bytes), no flags
    g1[1] = (int)((ne & 0xffffu) << 16);     // tensor_dim0[15:0]   (bits 63:48)
    g1[2] = (int)(((ne >> 16) & 0xffffu)     // tensor_dim0[31:16]  (bits 79:64)
                  | (1u << 16));             // tensor_dim1 = 1     (bits 95:80)
    g1[3] = (int)((ne & 0xffffu) << 16);     // tile_dim0 = nelem   (bits 127:112)
    g1[4] = 1;                               // tile_dim1 = 1, tile_dim2 = 0
    g1[5] = (int)ne;                         // tensor_dim0_stride[31:0]
    g1[6] = 0;                               // stride hi / dim1_stride lo
    g1[7] = 0;
    v4i z4 = {0, 0, 0, 0};
    v8i z8 = {0, 0, 0, 0, 0, 0, 0, 0};
    // 6-arg flavor (clang-23 / therock headers): (g0, g1, g2, g3, g4, cpol)
    __builtin_amdgcn_tensor_load_to_lds(g0, g1, z4, z4, z8, 0);
}

// ---------------------------------------------------------------------------
// Generic conv1d (k=3, zero pad) as WMMA GEMM.
//   y[b,co,p] = m[b,p] * sum_{ci,k} Wf[co,ci,k] * x[b,ci,p+k-1]
// transposed=0:  Wf[co,ci,k] = w[(co*Cin+ci)*3 + k]        (w shape (Cout,Cin,3))
// transposed=1:  Wf[co,ci,k] = w[(ci*Cout+co)*3 + (2-k)]   (w shape (Cin,Cout,3))
// GEMM K index: kk = k*Cpad + ci, Cpad = ceil16(Cin) in {32,64,128}; every
// 16-wide fragment run stays inside one k segment -> contiguous LDS loads.
// grid = (L/PTILE, Cout/16, NB), block = 256 (8 waves, one 16x16 tile each).
// ---------------------------------------------------------------------------
__global__ __launch_bounds__(256)
void conv1d_wmma_kernel(const float* __restrict__ x, const float* __restrict__ w,
                        const float* __restrict__ m, float* __restrict__ y,
                        int Cin, int Cout, int L, int transposed)
{
    __shared__ __align__(32) _Float16 sXt[(PTILE + 2) * 144];  // [t][ci], stride Cpad+16
    __shared__ __align__(32) _Float16 sW[16 * 384];            // [row][kk], stride 3*Cpad

    const int b   = blockIdx.z;
    const int co0 = blockIdx.y * 16;
    const int p0  = blockIdx.x * PTILE;
    const int tid = threadIdx.x;

    const int Cpad = (Cin + 15) & ~15;      // 32 / 64 / 128
    const int csh  = __builtin_ctz(Cpad);
    const int KP   = 3 * Cpad;              // 96 / 192 / 384 (multiples of 32)
    const int SBS  = Cpad + 16;             // multiple of 16 halves (32B rows)

    // stage input tile transposed (coalesced reads along positions)
    const float* xb = x + (size_t)b * Cin * L;
    for (int idx = tid; idx < Cpad * (PTILE + 2); idx += 256) {
        int ci = idx / (PTILE + 2);
        int t  = idx - ci * (PTILE + 2);
        int gp = p0 + t - 1;
        float v = (ci < Cin && gp >= 0 && gp < L) ? xb[(size_t)ci * L + gp] : 0.0f;
        sXt[t * SBS + ci] = (_Float16)v;
    }
    // stage 16-row weight panel in kk = k*Cpad + ci order (zero-padded ci)
    for (int idx = tid; idx < 16 * KP; idx += 256) {
        int r  = idx / KP;
        int kk = idx - r * KP;
        int k  = kk >> csh;
        int ci = kk & (Cpad - 1);
        float v = 0.0f;
        if (ci < Cin) {
            int co = co0 + r;
            v = transposed ? w[((size_t)ci * Cout + co) * 3 + (2 - k)]
                           : w[((size_t)co * Cin + ci) * 3 + k];
        }
        sW[r * KP + kk] = (_Float16)v;
    }
    __syncthreads();

    const int wave = tid >> 5, lane = tid & 31;
    const int half = lane >> 4, mn = lane & 15;
    const int col  = wave * 16 + mn;        // position within block tile

    v8f acc = {0.f, 0.f, 0.f, 0.f, 0.f, 0.f, 0.f, 0.f};
    const _Float16* wrow = &sW[mn * KP];

    for (int kc = 0; kc < KP; kc += 32) {
        // A fragment: 16 contiguous halves (32B-aligned)
        v16h afrag = *(const v16h*)(wrow + kc + half * 16);
        // B fragment: 16 contiguous halves (32B-aligned)
        int Kst = kc + half * 16;
        int k   = Kst >> csh;
        int cio = Kst & (Cpad - 1);
        v16h bfrag = *(const v16h*)&sXt[(col + k) * SBS + cio];

        acc = __builtin_amdgcn_wmma_f32_16x16x32_f16(
                  false, afrag, false, bfrag, (short)0, acc, false, false);
    }

    const float* mb = m + (size_t)b * L;
    float* yb = y + (size_t)b * Cout * L;
    const int p = p0 + col;
    const float mm = mb[p];
#pragma unroll
    for (int r = 0; r < 8; ++r) {
        int co = co0 + r + 8 * half;        // C/D layout: VGPR r -> M = r + 8*half
        yb[(size_t)co * L + p] = acc[r] * mm;
    }
}

// ---------------------------------------------------------------------------
// Masked instance norm (+optional ReLU, +optional skip add), in place.
// Row + mask are DMA'd into LDS by the Tensor Data Mover, then the three
// passes run against LDS.  grid = (C, NB), block = 256.
// ---------------------------------------------------------------------------
__global__ __launch_bounds__(256)
void inorm_kernel(float* __restrict__ xc, const float* __restrict__ m,
                  const float* __restrict__ skip, int C, int L, int do_relu)
{
    __shared__ __align__(16) float srow[NFULL];
    __shared__ __align__(16) float smask[NFULL];
    __shared__ float redA[256];
    __shared__ float redB[256];

    const int b = blockIdx.y, c = blockIdx.x, tid = threadIdx.x;
    float* row = xc + ((size_t)b * C + c) * L;
    const float* mrow = m + (size_t)b * L;

    if (tid < 32) {                         // wave 0 issues the TDM copies
        tdm_load_row_f32(lds_off_u32(srow), row, L);
        tdm_load_row_f32(lds_off_u32(smask), mrow, L);
        __builtin_amdgcn_s_wait_tensorcnt(0);
    }
    __syncthreads();

    float sx = 0.f, sm = 0.f;
    for (int i = tid; i < L; i += 256) { float mm = smask[i]; sx += srow[i] * mm; sm += mm; }
    redA[tid] = sx; redB[tid] = sm; __syncthreads();
    for (int s = 128; s > 0; s >>= 1) {
        if (tid < s) { redA[tid] += redA[tid + s]; redB[tid] += redB[tid + s]; }
        __syncthreads();
    }
    const float denom = redB[0];
    const float mean  = redA[0] / denom;
    __syncthreads();

    float sv = 0.f;
    for (int i = tid; i < L; i += 256) {
        float mm = smask[i];
        float d  = (srow[i] - mean * mm) * mm;
        sv += d * d;
    }
    redA[tid] = sv; __syncthreads();
    for (int s = 128; s > 0; s >>= 1) {
        if (tid < s) redA[tid] += redA[tid + s];
        __syncthreads();
    }
    const float inv = rsqrtf(redA[0] / denom + 1e-5f);
    const float* srowskip = skip ? skip + ((size_t)b * C + c) * L : nullptr;
    for (int i = tid; i < L; i += 256) {
        float mm = smask[i];
        float v  = (srow[i] - mean * mm) * inv;
        if (do_relu) v = fmaxf(v, 0.f);
        if (srowskip) v += srowskip[i];
        row[i] = v;
    }
}

// ---------------------------------------------------------------------------
// Elementwise helpers
// ---------------------------------------------------------------------------
__global__ void axpy_kernel(float* __restrict__ x, const float* __restrict__ y, int n)
{
    int i = blockIdx.x * blockDim.x + threadIdx.x;
    if (i < n) x[i] = x[i] - HSTEP * y[i];
}

__global__ void copy_kernel(float* __restrict__ d, const float* __restrict__ s, int n)
{
    int i = blockIdx.x * blockDim.x + threadIdx.x;
    if (i < n) d[i] = s[i];
}

__global__ void pool_mask_kernel(const float* __restrict__ m, float* __restrict__ mo,
                                 int Li, int Lo)
{
    int idx = blockIdx.x * blockDim.x + threadIdx.x;
    if (idx >= NB * Lo) return;
    int b = idx / Lo, i = idx - b * Lo;
    float s = 0.f;
    int base = 2 * i - 1;
#pragma unroll
    for (int t = 0; t < 3; ++t) {
        int p = base + t;
        if (p >= 0 && p < Li) s += m[(size_t)b * Li + p];
    }
    mo[idx] = ((s / 3.f) >= 0.5f) ? 1.0f : 0.0f;
}

__global__ void pool_x_kernel(const float* __restrict__ x, const float* __restrict__ mo,
                              float* __restrict__ y, int C, int Li, int Lo)
{
    int idx = blockIdx.x * blockDim.x + threadIdx.x;
    if (idx >= NB * C * Lo) return;
    int i  = idx % Lo;
    int bc = idx / Lo;
    int b  = bc / C;
    float s = 0.f;
    int base = 2 * i - 1;
#pragma unroll
    for (int t = 0; t < 3; ++t) {
        int p = base + t;
        if (p >= 0 && p < Li) s += x[(size_t)bc * Li + p];
    }
    y[(size_t)bc * Lo + i] = (s / 3.f) * mo[(size_t)b * Lo + i];
}

__global__ void upsample_kernel(const float* __restrict__ x, const float* __restrict__ m,
                                float* __restrict__ y, int C, int Li, int Lo)
{
    int idx = blockIdx.x * blockDim.x + threadIdx.x;
    if (idx >= NB * C * Lo) return;
    int p  = idx % Lo;
    int bc = idx / Lo;
    int b  = bc / C;
    y[(size_t)bc * Lo + p] = x[(size_t)bc * Li + (p >> 1)] * m[(size_t)b * Lo + p];
}

// final 1x1 conv (32 -> 3) + clamp to fixed coordinates where flag==1
__global__ void head_kernel(const float* __restrict__ xc, const float* __restrict__ W,
                            const float* __restrict__ m, const float* __restrict__ xin,
                            float* __restrict__ X, int L)
{
    int idx = blockIdx.x * blockDim.x + threadIdx.x;
    if (idx >= NB * L) return;
    int b = idx / L, p = idx - b * L;
    float mm   = m[idx];
    float flag = xin[((size_t)b * 24 + 23) * L + p];
#pragma unroll
    for (int d = 0; d < 3; ++d) {
        float acc = 0.f;
#pragma unroll
        for (int ci = 0; ci < 32; ++ci)
            acc += W[d * 32 + ci] * xc[((size_t)b * 32 + ci) * L + p];
        float v = acc * mm;
        if (flag == 1.0f) v = xin[((size_t)b * 24 + 20 + d) * L + p];
        X[((size_t)b * 3 + d) * L + p] = v;
    }
}

// normalized bond vectors for the distance constraint
__global__ void dxn_kernel(const float* __restrict__ X, const float* __restrict__ m,
                           float* __restrict__ DXN, int L)
{
    int idx = blockIdx.x * blockDim.x + threadIdx.x;
    if (idx >= NB * (L - 1)) return;
    int b = idx / (L - 1), j = idx - b * (L - 1);
    float dx[3], dsum = 0.f;
#pragma unroll
    for (int d = 0; d < 3; ++d) {
        float a = X[((size_t)b * 3 + d) * L + j + 1] - X[((size_t)b * 3 + d) * L + j];
        dx[d] = a; dsum += a * a;
    }
    float m0 = m[(size_t)b * L + j], m1 = m[(size_t)b * L + j + 1];
    bool avM = floorf((m0 + m1) * 0.5f) < 0.5f;
    float den   = sqrtf(dsum + (avM ? 1.f : 0.f));
    float scale = avM ? 0.f : DC_;
#pragma unroll
    for (int d = 0; d < 3; ++d)
        DXN[((size_t)b * 3 + d) * (L - 1) + j] = dx[d] / den * scale;
}

// serial left/right scans (inherently sequential in j); one lane per batch
__global__ void scan_kernel(const float* __restrict__ X, const float* __restrict__ DXN,
                            const float* __restrict__ xin,
                            float* __restrict__ XL, float* __restrict__ XR,
                            float* __restrict__ WL, float* __restrict__ WR, int L)
{
    int b = blockIdx.x;
    if (threadIdx.x != 0) return;
    const float* fx  = xin + ((size_t)b * 24 + 23) * L;
    const float* Xb  = X   + (size_t)b * 3 * L;
    const float* Db  = DXN + (size_t)b * 3 * (L - 1);
    float* XLb = XL + (size_t)b * 3 * L;
    float* XRb = XR + (size_t)b * 3 * L;
    float* WLb = WL + (size_t)b * L;
    float* WRb = WR + (size_t)b * L;
    const int Ld = L - 1;

    // forward
    bool  f = (fx[0] == 1.0f);
    float p0 = Xb[0], p1 = Xb[L], p2 = Xb[2 * L];
    XLb[0] = p0; XLb[L] = p1; XLb[2 * L] = p2;
    float wl = f ? 0.f : BIG_;
    WLb[0] = wl;
    bool seen = f;
    for (int j = 1; j < L; ++j) {
        bool  fj = (fx[j] == 1.0f);
        float x0 = Xb[j], x1 = Xb[L + j], x2 = Xb[2 * L + j];
        float v0, v1, v2;
        if (fj || !seen) { v0 = x0; v1 = x1; v2 = x2; }
        else { v0 = p0 + Db[j - 1]; v1 = p1 + Db[Ld + j - 1]; v2 = p2 + Db[2 * Ld + j - 1]; }
        wl = fj ? 0.f : (wl + 1.f);
        XLb[j] = v0; XLb[L + j] = v1; XLb[2 * L + j] = v2; WLb[j] = wl;
        p0 = v0; p1 = v1; p2 = v2; seen = seen || fj;
    }
    // backward
    f = (fx[L - 1] == 1.0f);
    p0 = Xb[L - 1]; p1 = Xb[2 * L - 1]; p2 = Xb[3 * L - 1];
    XRb[L - 1] = p0; XRb[2 * L - 1] = p1; XRb[3 * L - 1] = p2;
    float wr = f ? 0.f : BIG_;
    WRb[L - 1] = wr;
    seen = f;
    for (int j = L - 2; j >= 0; --j) {
        bool  fj = (fx[j] == 1.0f);
        float x0 = Xb[j], x1 = Xb[L + j], x2 = Xb[2 * L + j];
        float v0, v1, v2;
        if (fj || !seen) { v0 = x0; v1 = x1; v2 = x2; }
        else { v0 = p0 - Db[j]; v1 = p1 - Db[Ld + j]; v2 = p2 - Db[2 * Ld + j]; }
        wr = fj ? 0.f : (wr + 1.f);
        XRb[j] = v0; XRb[L + j] = v1; XRb[2 * L + j] = v2; WRb[j] = wr;
        p0 = v0; p1 = v1; p2 = v2; seen = seen || fj;
    }
}

__global__ void combine_kernel(const float* __restrict__ XL, const float* __restrict__ XR,
                               const float* __restrict__ WL, const float* __restrict__ WR,
                               const float* __restrict__ m, float* __restrict__ out, int L)
{
    int idx = blockIdx.x * blockDim.x + threadIdx.x;
    if (idx >= NB * L) return;
    int b = idx / L, i = idx - b * L;
    float wl = WL[idx], wr = WR[idx], ws = wl + wr;
    float cl, cr;
    if (ws == 0.f) { cl = 0.5f; cr = 0.5f; }
    else           { cl = wr / ws; cr = wl / ws; }
    float mm = m[idx];
#pragma unroll
    for (int d = 0; d < 3; ++d) {
        size_t o = ((size_t)b * 3 + d) * L + i;
        out[o] = (XL[o] * cl + XR[o] * cr) * mm;
    }
}

__global__ void mean_kernel(const float* __restrict__ X, float* __restrict__ MEAN, int L)
{
    __shared__ float red[256];
    int b = blockIdx.x, tid = threadIdx.x;
    for (int d = 0; d < 3; ++d) {
        float s = 0.f;
        for (int i = tid; i < L; i += 256) s += X[((size_t)b * 3 + d) * L + i];
        red[tid] = s; __syncthreads();
        for (int st = 128; st > 0; st >>= 1) {
            if (tid < st) red[tid] += red[tid + st];
            __syncthreads();
        }
        if (tid == 0) MEAN[b * 3 + d] = red[0] / (float)L;
        __syncthreads();
    }
}

__global__ void center_kernel(const float* __restrict__ X, const float* __restrict__ MEAN,
                              float* __restrict__ ZC, float* __restrict__ SQ, int L)
{
    int idx = blockIdx.x * blockDim.x + threadIdx.x;
    if (idx >= NB * L) return;
    int b = idx / L, i = idx - b * L;
    float sq = 0.f;
#pragma unroll
    for (int d = 0; d < 3; ++d) {
        float z = X[((size_t)b * 3 + d) * L + i] - MEAN[b * 3 + d];
        ZC[((size_t)b * L + i) * 3 + d] = z;
        sq += z * z;
    }
    SQ[idx] = sq;
}

// pairwise distance output: 134 MB -> the HBM-bound tail (~6 us at 23.3 TB/s)
__global__ __launch_bounds__(256)
void dist_kernel(const float* __restrict__ ZC, const float* __restrict__ SQ,
                 float* __restrict__ D, int L)
{
    int b = blockIdx.z;
    int i = blockIdx.y * 16 + threadIdx.y;
    int j = blockIdx.x * 16 + threadIdx.x;
    const float* zb = ZC + (size_t)b * L * 3;
    float zi0 = zb[(size_t)i * 3 + 0], zi1 = zb[(size_t)i * 3 + 1], zi2 = zb[(size_t)i * 3 + 2];
    float zj0 = zb[(size_t)j * 3 + 0], zj1 = zb[(size_t)j * 3 + 1], zj2 = zb[(size_t)j * 3 + 2];
    float dot = zi0 * zj0 + zi1 * zj1 + zi2 * zj2;
    float d2  = SQ[(size_t)b * L + i] + SQ[(size_t)b * L + j] - 2.f * dot;
    D[(size_t)b * L * L + (size_t)i * L + j] = sqrtf(fmaxf(d2, 0.f) + 1e-8f);
}

// ---------------------------------------------------------------------------
// Orchestration
// ---------------------------------------------------------------------------
extern "C" void kernel_launch(void* const* d_in, const int* in_sizes, int n_in,
                              void* d_out, int out_size, void* d_ws, size_t ws_size,
                              hipStream_t stream)
{
    (void)in_sizes; (void)n_in; (void)out_size; (void)ws_size;

    const float* x_in = (const float*)d_in[0];   // (8,24,2048)
    const float* m0   = (const float*)d_in[1];   // (8,2048)
    const float* K0   = (const float*)d_in[2];   // (32,24,3)
    const float* K1   = (const float*)d_in[3];   // (32,32,3)
    const float* K2   = (const float*)d_in[4];   // (32,32,3)
    const float* K3   = (const float*)d_in[5];   // (64,32,3)
    const float* K4   = (const float*)d_in[6];   // (64,64,3)
    const float* K5   = (const float*)d_in[7];   // (64,64,3)
    const float* K6   = (const float*)d_in[8];   // (128,64,3)
    const float* Wh   = (const float*)d_in[9];   // (3,32,1)

    const int n = NFULL;
    float* dists = (float*)d_out;                         // (8,2048,2048)
    float* outX  = dists + (size_t)NB * n * n;            // (8,3,2048)

    // workspace carve-up (floats)
    float* ws = (float*)d_ws;
    const size_t ACT = (size_t)NB * 128 * n;              // 2.10M floats
    float* A   = ws;               ws += ACT;
    float* B   = ws;               ws += ACT;
    float* Z   = ws;               ws += ACT;
    float* S0  = ws;               ws += (size_t)NB * 32 * n;        // skip @2048
    float* S1  = ws;               ws += (size_t)NB * 64 * (n / 2);  // skip @1024
    float* M1  = ws;               ws += (size_t)NB * (n / 2);
    float* M2  = ws;               ws += (size_t)NB * (n / 4);
    float* DXN = ws;               ws += (size_t)NB * 3 * n;
    float* XL  = ws;               ws += (size_t)NB * 3 * n;
    float* XR  = ws;               ws += (size_t)NB * 3 * n;
    float* WLb = ws;               ws += (size_t)NB * n;
    float* WRb = ws;               ws += (size_t)NB * n;
    float* XC  = ws;               ws += (size_t)NB * 3 * n;
    float* ZC  = ws;               ws += (size_t)NB * n * 3;
    float* SQ  = ws;               ws += (size_t)NB * n;
    float* MEANb = ws;             ws += (size_t)NB * 3;

    auto conv = [&](const float* xi, const float* w, const float* mm, float* y,
                    int Cin, int Cout, int L, int tr) {
        dim3 g(L / PTILE, Cout / 16, NB);
        conv1d_wmma_kernel<<<g, 256, 0, stream>>>(xi, w, mm, y, Cin, Cout, L, tr);
    };
    auto inorm = [&](float* xc, const float* mm, const float* skip, int C, int L, int relu) {
        dim3 g(C, NB);
        inorm_kernel<<<g, 256, 0, stream>>>(xc, mm, skip, C, L, relu);
    };
    auto ew = [&](int count) { return dim3(ceil_div_i(count, 256)); };

    // ---------------- encoder ----------------
    // K0
    conv(x_in, K0, m0, A, 24, 32, n, 0);
    inorm(A, m0, nullptr, 32, n, 1);
    // K1, K2 residual @2048 (32ch)
    const float* Kres32[2] = {K1, K2};
    for (int l = 0; l < 2; ++l) {
        conv(A, Kres32[l], m0, Z, 32, 32, n, 0);
        inorm(Z, m0, nullptr, 32, n, 1);
        conv(Z, Kres32[l], m0, B, 32, 32, n, 1);
        axpy_kernel<<<ew(NB * 32 * n), 256, 0, stream>>>(A, B, NB * 32 * n);
    }
    // K3 downsample: save skip, conv, norm, pool
    copy_kernel<<<ew(NB * 32 * n), 256, 0, stream>>>(S0, A, NB * 32 * n);
    conv(A, K3, m0, B, 32, 64, n, 0);
    inorm(B, m0, nullptr, 64, n, 1);
    pool_mask_kernel<<<ew(NB * (n / 2)), 256, 0, stream>>>(m0, M1, n, n / 2);
    pool_x_kernel<<<ew(NB * 64 * (n / 2)), 256, 0, stream>>>(B, M1, A, 64, n, n / 2);
    // K4, K5 residual @1024 (64ch)
    const float* Kres64[2] = {K4, K5};
    for (int l = 0; l < 2; ++l) {
        conv(A, Kres64[l], M1, Z, 64, 64, n / 2, 0);
        inorm(Z, M1, nullptr, 64, n / 2, 1);
        conv(Z, Kres64[l], M1, B, 64, 64, n / 2, 1);
        axpy_kernel<<<ew(NB * 64 * (n / 2)), 256, 0, stream>>>(A, B, NB * 64 * (n / 2));
    }
    // K6 downsample
    copy_kernel<<<ew(NB * 64 * (n / 2)), 256, 0, stream>>>(S1, A, NB * 64 * (n / 2));
    conv(A, K6, M1, B, 64, 128, n / 2, 0);
    inorm(B, M1, nullptr, 128, n / 2, 1);
    pool_mask_kernel<<<ew(NB * (n / 4)), 256, 0, stream>>>(M1, M2, n / 2, n / 4);
    pool_x_kernel<<<ew(NB * 128 * (n / 4)), 256, 0, stream>>>(B, M2, A, 128, n / 2, n / 4);

    // ---------------- decoder ----------------
    // i=6: upsample, convT(K6) 128->64, norm + relu + skip S1
    upsample_kernel<<<ew(NB * 128 * (n / 2)), 256, 0, stream>>>(A, M1, B, 128, n / 4, n / 2);
    conv(B, K6, M1, Z, 128, 64, n / 2, 1);
    inorm(Z, M1, S1, 64, n / 2, 1);
    // i=5,4: residual (convT first) @1024 (64ch), xc in Z
    const float* KresD64[2] = {K5, K4};
    for (int l = 0; l < 2; ++l) {
        conv(Z, KresD64[l], M1, A, 64, 64, n / 2, 1);
        inorm(A, M1, nullptr, 64, n / 2, 1);
        conv(A, KresD64[l], M1, B, 64, 64, n / 2, 0);
        axpy_kernel<<<ew(NB * 64 * (n / 2)), 256, 0, stream>>>(Z, B, NB * 64 * (n / 2));
    }
    // i=3: upsample, convT(K3) 64->32, norm + relu + skip S0
    upsample_kernel<<<ew(NB * 64 * n), 256, 0, stream>>>(Z, m0, A, 64, n / 2, n);
    conv(A, K3, m0, B, 64, 32, n, 1);
    inorm(B, m0, S0, 32, n, 1);
    // i=2,1: residual @2048 (32ch), xc in B
    const float* KresD32[2] = {K2, K1};
    for (int l = 0; l < 2; ++l) {
        conv(B, KresD32[l], m0, A, 32, 32, n, 1);
        inorm(A, m0, nullptr, 32, n, 1);
        conv(A, KresD32[l], m0, Z, 32, 32, n, 0);
        axpy_kernel<<<ew(NB * 32 * n), 256, 0, stream>>>(B, Z, NB * 32 * n);
    }

    // ---------------- head + distance constraint ----------------
    head_kernel<<<ew(NB * n), 256, 0, stream>>>(B, Wh, m0, x_in, XC, n);
    dxn_kernel<<<ew(NB * (n - 1)), 256, 0, stream>>>(XC, m0, DXN, n);
    scan_kernel<<<dim3(NB), 32, 0, stream>>>(XC, DXN, x_in, XL, XR, WLb, WRb, n);
    combine_kernel<<<ew(NB * n), 256, 0, stream>>>(XL, XR, WLb, WRb, m0, outX, n);

    // ---------------- pairwise distances ----------------
    mean_kernel<<<dim3(NB), 256, 0, stream>>>(outX, MEANb, n);
    center_kernel<<<ew(NB * n), 256, 0, stream>>>(outX, MEANb, ZC, SQ, n);
    dist_kernel<<<dim3(n / 16, n / 16, NB), dim3(16, 16), 0, stream>>>(ZC, SQ, dists, n);
}
